// BidirectionalAdditiveAttention_41686952575660
// MI455X (gfx1250) — compile-verified, matrix operand order checked by reading, not run on hardware
//
#include <hip/hip_runtime.h>
#include <hip/hip_bf16.h>

// ---------------------------------------------------------------------------
// BidirectionalAdditiveAttention for MI455X (gfx1250, wave32, WMMA + TDM)
//
// Sizes (fixed): B=8, L1=L2=512, K1D=K2D=256, A=128, DV=256
//
// Pipeline:
//   1) gemm_bf16_wmma : p1 = k1@W1+b1, p2 = k2@W2+b2   (LDS-staged, 4 WMMA/wave/K-step)
//   2) score_kernel   : fused tanh-score + mask        (p1/p2 tiles via tensor_load_to_lds)
//   3) softmax_kernel : w2 = softmax_j(score), w1 = softmax_i(score^T)
//   4) gemm_bf16_wmma : o2 = w2@v2, o1 = w1@v1
// ---------------------------------------------------------------------------

typedef __attribute__((ext_vector_type(16))) __bf16        v16bf;
typedef __attribute__((ext_vector_type(8)))  float         v8f;
typedef __attribute__((ext_vector_type(4)))  float         f32x4;
typedef __attribute__((ext_vector_type(4)))  unsigned int  uint32x4;
typedef __attribute__((ext_vector_type(2)))  unsigned int  uint32x2;
typedef __attribute__((ext_vector_type(8)))  int           int32x8;
typedef __attribute__((ext_vector_type(4)))  int           int32x4;

#define BATCH   8
#define LSEQ    512
#define KDIM    256
#define ADIM    128
#define DVDIM   256

// GEMM tiling
#define BM      128
#define BN      64
#define BK      32
#define LDPAD   40       // ushort row stride in LDS (80B): conflict-free b128 frags

static __device__ inline unsigned short bf16_trunc(float f) {
    return (unsigned short)(__builtin_bit_cast(unsigned int, f) >> 16);
}
// pack two f32 -> one dword of two bf16 (truncation; 2 VALU)
static __device__ inline unsigned int bf16_pack2(float lo, float hi) {
    return (__builtin_bit_cast(unsigned int, hi) & 0xFFFF0000u) |
           (__builtin_bit_cast(unsigned int, lo) >> 16);
}

union bfrag {
    uint32x4 q[2];
    v16bf    v;
};

// ---------------------------------------------------------------------------
// C(MxN) = A(MxK) @ B(KxN) [+bias], row-major f32, bf16 WMMA accumulation.
// Block: 256 threads = 8 waves; block tile 128x64; wave w -> rows [16w,16w+16)
// across all 64 cols (4 accumulators). K stepped by 32 with LDS staging.
//
// LDS fragment layout matches ISA 7.12.2 16-bit A/B:
//   lane L: row/col = L&15, khalf = L>>4;
//   elements 0..7  -> K = khalf*8 + e        (16 contiguous bytes -> ds_load_b128)
//   elements 8..15 -> K = 16 + khalf*8 + e-8 (16 contiguous bytes -> ds_load_b128)
// ---------------------------------------------------------------------------
__global__ void gemm_bf16_wmma(const float* __restrict__ A, long long strideA,
                               const float* __restrict__ Bm, long long strideB,
                               const float* __restrict__ bias,
                               float* __restrict__ C, long long strideC,
                               int M, int N, int K)
{
    __shared__ unsigned short sA[BM][LDPAD];   // 128 x 32 bf16 (padded)
    __shared__ unsigned short sB[BN][LDPAD];   // 64 x 32 bf16, K-contiguous (padded)

    const int batch = blockIdx.z;
    A  += (long long)batch * strideA;
    Bm += (long long)batch * strideB;
    C  += (long long)batch * strideC;

    const int row0 = blockIdx.x * BM;
    const int col0 = blockIdx.y * BN;
    const int tid  = threadIdx.x;          // 0..255
    const int w    = tid >> 5;             // wave 0..7
    const int lane = tid & 31;
    const int mn   = lane & 15;
    const int kh   = lane >> 4;

    v8f acc[4] = {{}, {}, {}, {}};

    for (int k0 = 0; k0 < K; k0 += BK) {
        // ---- cooperative stage A tile: 128x32 f32 -> bf16, coalesced b128 ----
        const float* Abase = A + (long long)row0 * K + k0;
#pragma unroll
        for (int p = 0; p < 4; ++p) {
            const int idx = p * 256 + tid;        // 0..1023 float4s
            const int r   = idx >> 3;             // row 0..127
            const int c4  = idx & 7;              // float4 col 0..7
            f32x4 f = *(const f32x4*)(Abase + (long long)r * K + c4 * 4);
            uint32x2 pk = { bf16_pack2(f[0], f[1]), bf16_pack2(f[2], f[3]) };
            *(uint32x2*)&sA[r][c4 * 4] = pk;
        }
        // ---- cooperative stage B tile: 32x64 f32 -> bf16 transposed (K-contig) ----
        const float* Bbase = Bm + (long long)k0 * N + col0;
#pragma unroll
        for (int p = 0; p < 2; ++p) {
            const int idx = p * 256 + tid;        // 0..511 float4s
            const int r   = idx >> 4;             // K row 0..31
            const int c4  = idx & 15;             // float4 col 0..15
            f32x4 f = *(const f32x4*)(Bbase + (long long)r * N + c4 * 4);
#pragma unroll
            for (int q = 0; q < 4; ++q)
                sB[c4 * 4 + q][r] = bf16_trunc(f[q]);
        }
        __syncthreads();

        // prefetch next K panels while WMMAs run (global_prefetch_b8)
        if (k0 + BK < K) {
            __builtin_prefetch(Abase + BK + (long long)(tid & (BM - 1)) * K, 0, 1);
            __builtin_prefetch(Bbase + (long long)BK * N + (tid & (BK - 1)) * (long long)N, 0, 1);
        }

        // ---- fragments from LDS (2 x ds_load_b128 each) + 4 WMMAs ----
        bfrag a;
        a.q[0] = *(const uint32x4*)&sA[(w << 4) + mn][kh * 8];
        a.q[1] = *(const uint32x4*)&sA[(w << 4) + mn][16 + kh * 8];
#pragma unroll
        for (int c = 0; c < 4; ++c) {
            bfrag b;
            b.q[0] = *(const uint32x4*)&sB[(c << 4) + mn][kh * 8];
            b.q[1] = *(const uint32x4*)&sB[(c << 4) + mn][16 + kh * 8];
            acc[c] = __builtin_amdgcn_wmma_f32_16x16x32_bf16(false, a.v, false, b.v,
                                                             (short)0, acc[c], false, false);
        }
        __syncthreads();
    }

    // ---- epilogue: D layout VGPR v, lane L -> M = v + (L>=16)*8, N = L&15 ----
#pragma unroll
    for (int c = 0; c < 4; ++c) {
#pragma unroll
        for (int v = 0; v < 8; ++v) {
            const int m = row0 + (w << 4) + v + kh * 8;
            const int n = col0 + (c << 4) + mn;
            float out = acc[c][v];
            if (bias) out += bias[n];
            C[(long long)m * N + n] = out;
        }
    }
}

// ---------------------------------------------------------------------------
// Fused additive-attention score with TDM tile staging:
//   score[b,i,j] = sum_a ws[a]*tanh(p1[b,i,a] + p2[b,j,a]) + bs,
//   masked ((i>=len1[b]) XOR (j>=len2[b])) -> -inf.
// Block = 256 threads -> 16x16 (i,j) tile. p1/p2 16x128 f32 tiles are DMA'd
// into LDS by the Tensor Data Mover (tensor_load_to_lds, TENSORcnt).
// ---------------------------------------------------------------------------
__global__ void score_kernel(const float* __restrict__ p1,
                             const float* __restrict__ p2,
                             const float* __restrict__ wsv,
                             const float* __restrict__ bs,
                             const int* __restrict__ len1,
                             const int* __restrict__ len2,
                             float* __restrict__ score)
{
    __shared__ float s_p1[16][ADIM];
    __shared__ float s_p2[16][ADIM];
    __shared__ float s_ws[ADIM];

    const int b  = blockIdx.z;
    const int i0 = blockIdx.x << 4;
    const int j0 = blockIdx.y << 4;
    const int tid = threadIdx.x;

    if (tid < 32) {  // wave 0 (uniform branch) issues the two TDM loads
        const unsigned lds1 = (unsigned)(unsigned long long)&s_p1[0][0];
        const unsigned lds2 = (unsigned)(unsigned long long)&s_p2[0][0];
        const unsigned long long ga1 =
            (unsigned long long)(p1 + ((long long)b * LSEQ + i0) * ADIM);
        const unsigned long long ga2 =
            (unsigned long long)(p2 + ((long long)b * LSEQ + j0) * ADIM);

        // D# group1 (shared by both): data_size=4B(code 2), tensor (4096 x 128) f32,
        // tile 128x16, dim0 stride 128. Bit-packing per ISA 8.4.
        const int32x8 g1 = {
            0x00020000,                      // workgroup_mask=0 | data_size=2<<16
            (int)(ADIM << 16),               // [63:48] = tensor_dim0 lo16 = 128
            (int)((BATCH * LSEQ) << 16),     // [95:80] = tensor_dim1 lo16 = 4096
            (int)(ADIM << 16),               // [127:112] = tile_dim0 = 128
            16,                              // [143:128] = tile_dim1 = 16
            ADIM,                            // [191:160] = tensor_dim0_stride = 128
            0, 0
        };
        const int32x4 gz4 = {0, 0, 0, 0};    // groups 2/3 unused (2D tile)
        const int32x8 gz8 = {0, 0, 0, 0, 0, 0, 0, 0};

        const uint32x4 g0a = { 1u, lds1, (unsigned)ga1,
                               (unsigned)((ga1 >> 32) & 0x1FFFFFFull) | (2u << 30) };
        const uint32x4 g0b = { 1u, lds2, (unsigned)ga2,
                               (unsigned)((ga2 >> 32) & 0x1FFFFFFull) | (2u << 30) };

        // 6-arg toolchain variant: (g0, g1, g2, g3, extra, cpol)
        __builtin_amdgcn_tensor_load_to_lds(g0a, g1, gz4, gz4, gz8, 0);
        __builtin_amdgcn_tensor_load_to_lds(g0b, g1, gz4, gz4, gz8, 0);
        __builtin_amdgcn_s_wait_tensorcnt(0);
    }
    if (tid < ADIM) s_ws[tid] = wsv[tid];
    __syncthreads();

    const int il = tid >> 4, jl = tid & 15;
    float acc = 0.0f;
#pragma unroll 8
    for (int a = 0; a < ADIM; ++a)
        acc += s_ws[a] * tanhf(s_p1[il][a] + s_p2[jl][a]);

    const int i = i0 + il, j = j0 + jl;
    const bool masked = (i >= len1[b]) != (j >= len2[b]);
    const float val = masked ? -__builtin_inff() : (acc + bs[0]);
    score[((long long)b * LSEQ + i) * LSEQ + j] = val;
}

// ---------------------------------------------------------------------------
// Row softmax over 512 entries. transposed==0: out[b,r,:] = softmax(score[b,r,:])
// transposed==1: out[b,r,c] = softmax over c of score[b,c,r].
// ---------------------------------------------------------------------------
__global__ void softmax_kernel(const float* __restrict__ score,
                               float* __restrict__ out, int transposed)
{
    __shared__ float red_max[8];
    __shared__ float red_sum[8];

    const int b = blockIdx.y;
    const int r = blockIdx.x;
    const float* sb = score + (long long)b * LSEQ * LSEQ;
    float* ob = out + ((long long)b * LSEQ + r) * LSEQ;

    const int tid = threadIdx.x;
    const int c0 = tid, c1 = tid + 256;

    const float x0 = transposed ? sb[(long long)c0 * LSEQ + r] : sb[(long long)r * LSEQ + c0];
    const float x1 = transposed ? sb[(long long)c1 * LSEQ + r] : sb[(long long)r * LSEQ + c1];

    float mx = fmaxf(x0, x1);
    for (int off = 16; off > 0; off >>= 1)
        mx = fmaxf(mx, __shfl_xor(mx, off, 32));
    if ((tid & 31) == 0) red_max[tid >> 5] = mx;
    __syncthreads();
    mx = red_max[0];
#pragma unroll
    for (int ww = 1; ww < 8; ++ww) mx = fmaxf(mx, red_max[ww]);

    const float e0 = __expf(x0 - mx);
    const float e1 = __expf(x1 - mx);
    float sm = e0 + e1;
    for (int off = 16; off > 0; off >>= 1)
        sm += __shfl_xor(sm, off, 32);
    if ((tid & 31) == 0) red_sum[tid >> 5] = sm;
    __syncthreads();
    sm = red_sum[0];
#pragma unroll
    for (int ww = 1; ww < 8; ++ww) sm += red_sum[ww];

    const float inv = 1.0f / sm;
    ob[c0] = e0 * inv;
    ob[c1] = e1 * inv;
}

// ---------------------------------------------------------------------------
// Launch
// ---------------------------------------------------------------------------
extern "C" void kernel_launch(void* const* d_in, const int* in_sizes, int n_in,
                              void* d_out, int out_size, void* d_ws, size_t ws_size,
                              hipStream_t stream)
{
    const float* k1   = (const float*)d_in[0];
    const float* k2   = (const float*)d_in[1];
    const float* v1   = (const float*)d_in[2];
    const float* v2   = (const float*)d_in[3];
    const float* W1   = (const float*)d_in[4];
    const float* b1   = (const float*)d_in[5];
    const float* W2   = (const float*)d_in[6];
    const float* b2   = (const float*)d_in[7];
    const float* wsv  = (const float*)d_in[8];
    const float* bs   = (const float*)d_in[9];
    const int*   len1 = (const int*)d_in[10];
    const int*   len2 = (const int*)d_in[11];

    // d_out: o1(B,L2,DV) | o2(B,L1,DV) | w1(B,L2,L1) | w2(B,L1,L2) | score(B,L1,L2)
    float* o1    = (float*)d_out;
    float* o2    = o1 + (long long)BATCH * LSEQ * DVDIM;
    float* w1    = o2 + (long long)BATCH * LSEQ * DVDIM;
    float* w2    = w1 + (long long)BATCH * LSEQ * LSEQ;
    float* score = w2 + (long long)BATCH * LSEQ * LSEQ;

    float* p1 = (float*)d_ws;
    float* p2 = p1 + (long long)BATCH * LSEQ * ADIM;

    const long long sAtt = (long long)LSEQ * LSEQ;
    const long long sVal = (long long)LSEQ * DVDIM;

    // 1) projections: (4096 x 256) @ (256 x 128) + bias
    {
        dim3 grid((BATCH * LSEQ) / BM, ADIM / BN, 1);
        gemm_bf16_wmma<<<grid, 256, 0, stream>>>(k1, 0, W1, 0, b1, p1, 0,
                                                 BATCH * LSEQ, ADIM, KDIM);
        gemm_bf16_wmma<<<grid, 256, 0, stream>>>(k2, 0, W2, 0, b2, p2, 0,
                                                 BATCH * LSEQ, ADIM, KDIM);
    }

    // 2) fused tanh score + masking (TDM tile staging)
    {
        dim3 grid(LSEQ / 16, LSEQ / 16, BATCH);
        score_kernel<<<grid, 256, 0, stream>>>(p1, p2, wsv, bs, len1, len2, score);
    }

    // 3) softmaxes
    {
        dim3 grid(LSEQ, BATCH);
        softmax_kernel<<<grid, 256, 0, stream>>>(score, w2, 0);
        softmax_kernel<<<grid, 256, 0, stream>>>(score, w1, 1);
    }

    // 4) outputs: batched (512 x 512) @ (512 x 256)
    {
        dim3 grid(LSEQ / BM, DVDIM / BN, BATCH);
        gemm_bf16_wmma<<<grid, 256, 0, stream>>>(w2, sAtt, v2, sVal, nullptr,
                                                 o2, sVal, LSEQ, DVDIM, LSEQ);
        gemm_bf16_wmma<<<grid, 256, 0, stream>>>(w1, sAtt, v1, sVal, nullptr,
                                                 o1, sVal, LSEQ, DVDIM, LSEQ);
    }
}
